// InteracionBlock_24927990186258
// MI455X (gfx1250) — matrix operand order repaired and checked by reading, not compiled, not run on hardware
//
#include <hip/hip_runtime.h>

// ---------------------------------------------------------------------------
// SchNet interaction block for MI455X (gfx1250), fully fused cfconv.
// B=8, S=256, D=256, N_RBF=300 (padded to 320), H=64.
// Heavy math: v_wmma_f32_16x16x32_bf16. Weight staging: Tensor Data Mover
// (tensor_load_to_lds) from pre-converted bf16 weight images.
// ---------------------------------------------------------------------------

#define LN2F 0.69314718056f

typedef __bf16 bf16;
typedef __bf16 v16bf __attribute__((ext_vector_type(16)));
typedef float  v8f   __attribute__((ext_vector_type(8)));
typedef unsigned int u32x4 __attribute__((ext_vector_type(4)));
typedef int          i32x4 __attribute__((ext_vector_type(4)));
typedef int          i32x8 __attribute__((ext_vector_type(8)));

union Frag32B { uint4 q[2]; v16bf v; };

__device__ __forceinline__ float sspf(float x) {
  // softplus(x) - ln2, stable form
  return fmaxf(x, 0.f) + __logf(1.f + __expf(-fabsf(x))) - LN2F;
}

// Low 32 bits of a generic pointer to a __shared__ object = LDS byte offset.
__device__ __forceinline__ unsigned ldsAddrOf(const void* p) {
  return (unsigned)(uintptr_t)p;
}

// ---------------------------------------------------------------------------
// TDM: load a 2D bf16 tile [tileH rows x tileW elems] (row stride rowStride
// elems in global) into contiguous LDS at ldsOff. D# layout per
// CDNA5 ISA. Issue from ONE wave only.
// Toolchain uses the 6-arg builtin form (extra int32x8 group, then cpol).
// ---------------------------------------------------------------------------
__device__ __forceinline__ void tdm_load_bf16(unsigned ldsOff, const bf16* gptr,
                                              unsigned tileW, unsigned tileH,
                                              unsigned rowStride) {
  const unsigned long long ga = (unsigned long long)(uintptr_t)gptr;
  u32x4 g0;
  g0[0] = 1u;                                         // count=1, no gather/restore
  g0[1] = ldsOff;                                     // lds_addr [63:32]
  g0[2] = (unsigned)(ga & 0xffffffffu);               // global_addr lo
  g0[3] = (unsigned)((ga >> 32) & 0x01ffffffu)        // global_addr [56:32]
          | (2u << 30);                               // type=2 ("image")
  const unsigned tensorW = 1u << 20;                  // generous dims: no OOB clip
  const unsigned tensorH = 1u << 20;
  i32x8 g1;
  g1[0] = (int)(1u << 16);                            // wg_mask=0, data_size=1 (2B)
  g1[1] = (int)((tensorW & 0xffffu) << 16);           // tensor_dim0 [15:0]
  g1[2] = (int)((tensorW >> 16) | ((tensorH & 0xffffu) << 16));
  g1[3] = (int)((tensorH >> 16) | (tileW << 16));     // tile_dim0
  g1[4] = (int)(tileH & 0xffffu);                     // tile_dim1 (tile_dim2=0)
  g1[5] = (int)rowStride;                             // tensor_dim0_stride lo
  g1[6] = 0;                                          // td0s hi | td1s lo
  g1[7] = 0;                                          // td1s hi
  const i32x4 z4 = {0, 0, 0, 0};                      // groups 2/3 unused (2D)
  const i32x8 z8 = {0, 0, 0, 0, 0, 0, 0, 0};
  __builtin_amdgcn_tensor_load_to_lds(g0, g1, z4, z4, z8, 0);
}

// A fragment (16x32 bf16) from LDS, row-major, rowStride in bf16 elems.
// ISA layout: lanes 0-15: M=lane, K bytes [0..15] & [32..47];
//             lanes 16-31: M=lane-16, K bytes [16..31] & [48..63].
__device__ __forceinline__ v16bf ldsFragA(const bf16* tile, int rowStride, int lane) {
  const int r = lane & 15;
  const int off = (lane & 16) ? 16 : 0;
  const char* p = (const char*)(tile + r * rowStride) + off;
  Frag32B f;
  f.q[0] = *(const uint4*)(p);
  f.q[1] = *(const uint4*)(p + 32);
  return f.v;
}

// B fragment (32x16 bf16) from LDS: N = colOff + lane%16, K = 8h+s (s<8), 16+8h+(s-8) (s>=8).
__device__ __forceinline__ v16bf ldsFragB(const bf16* tile, int rowStride, int colOff, int lane) {
  const bf16* p = tile + colOff + (lane & 15);
  const int h8 = (lane >> 4) * 8;
  v16bf v;
#pragma unroll
  for (int s = 0; s < 8; ++s) {
    v[s] = p[(h8 + s) * rowStride];
    v[8 + s] = p[(16 + h8 + s) * rowStride];
  }
  return v;
}

// ---------------------------------------------------------------------------
// One-time weight conversion f32 -> bf16 (Ws1 zero-padded 300 -> 320 rows).
// Grid: 256 blocks x 256 threads = 65536 threads.
// ---------------------------------------------------------------------------
__global__ void __launch_bounds__(256) k_prep(
    const float* __restrict__ Wa1, const float* __restrict__ Wa2,
    const float* __restrict__ Wa3, const float* __restrict__ Ws1,
    const float* __restrict__ Ws2,
    bf16* __restrict__ wa1b, bf16* __restrict__ wa2b, bf16* __restrict__ wa3b,
    bf16* __restrict__ ws1b, bf16* __restrict__ ws2b)
{
  const int idx = blockIdx.x * 256 + threadIdx.x;     // 0..65535
  wa1b[idx] = (bf16)Wa1[idx];
  wa2b[idx] = (bf16)Wa2[idx];
  wa3b[idx] = (bf16)Wa3[idx];
  if (idx < 16384) ws2b[idx] = (bf16)Ws2[idx];
  if (idx < 20480) {
    const int row = idx >> 6, col = idx & 63;
    ws1b[idx] = (bf16)((row < 300) ? Ws1[row * 64 + col] : 0.f);
  }
}

// ---------------------------------------------------------------------------
// Atom-wise layer: out = postmask( act( premask(A) @ W + bias ) )
// A: [2048 x 256] f32, Wb: [256 x 256] bf16, mask: [2048] f32.
// Block: 128 threads (4 waves); tile 16 rows x 64 cols; grid 512.
// Weight strip [256 x 64] staged to LDS via TDM.
// ---------------------------------------------------------------------------
__global__ void __launch_bounds__(128) k_dense256(
    const float* __restrict__ A, const bf16* __restrict__ Wb,
    const float* __restrict__ bias, const float* __restrict__ mask,
    float* __restrict__ out, int preMask, int doSsp, int postMask)
{
  __shared__ __align__(16) bf16 sW[256 * 64];          // 32 KB weight strip
  __shared__ __align__(16) bf16 sA[16 * 32];

  const int tid  = threadIdx.x;
  const int lane = tid & 31;
  const int wave = tid >> 5;                           // 0..3
  const int rowBase = (blockIdx.x >> 2) * 16;          // 128 row tiles
  const int strip = blockIdx.x & 3;                    // 64-col strip
  const int nt = strip * 4 + wave;                     // global column tile

  if (wave == 0) {
    tdm_load_bf16(ldsAddrOf(sW), Wb + strip * 64, 64, 256, 256);
    __builtin_amdgcn_s_wait_tensorcnt(0);
  }
  __syncthreads();

  const int arow = tid >> 3;                           // 0..15
  const int acol = (tid & 7) * 4;                      // 0..28
  const float mr = preMask ? mask[rowBase + arow] : 1.f;

  v8f acc = {};
#pragma unroll
  for (int kt = 0; kt < 8; ++kt) {
    const int k0 = kt * 32;
    float4 av = *(const float4*)(A + (rowBase + arow) * 256 + k0 + acol);
    sA[arow * 32 + acol + 0] = (bf16)(av.x * mr);
    sA[arow * 32 + acol + 1] = (bf16)(av.y * mr);
    sA[arow * 32 + acol + 2] = (bf16)(av.z * mr);
    sA[arow * 32 + acol + 3] = (bf16)(av.w * mr);
    __syncthreads();
    v16bf af = ldsFragA(sA, 32, lane);
    v16bf bf_ = ldsFragB(sW + k0 * 64, 64, wave * 16, lane);
    acc = __builtin_amdgcn_wmma_f32_16x16x32_bf16(false, af, false, bf_,
                                                  (short)0, acc, false, false);
    __syncthreads();
  }

  const int col = nt * 16 + (lane & 15);
  const float bv = bias[col];
#pragma unroll
  for (int r = 0; r < 8; ++r) {
    const int m = r + ((lane & 16) ? 8 : 0);
    const int row = rowBase + m;
    float x = acc[r] + bv;
    if (doSsp) x = sspf(x);
    if (postMask) x *= mask[row];
    out[row * 256 + col] = x;
  }
}

// ---------------------------------------------------------------------------
// Fused continuous-filter convolution. One block per (b,i); 256 threads = 8 waves.
// Both weight matrices live in LDS (TDM-staged bf16). Per chunk of 32 senders j:
//   rbf[32x320] bf16 -> WMMA (rbf @ Ws1) -> ssp -> h[32x64] bf16
//   -> WMMA (h @ Ws2) -> ssp -> fold with v1[b,j,d] into per-lane accumulators.
// Never materializes the [B,S,S,D] filter (512 MB) -> compute bound.
// ---------------------------------------------------------------------------
__global__ void __launch_bounds__(256) k_cfconv(
    const float* __restrict__ v1, const float* __restrict__ dist,
    const bf16* __restrict__ ws1b, const float* __restrict__ bs1,
    const bf16* __restrict__ ws2b, const float* __restrict__ bs2,
    float* __restrict__ u)
{
  __shared__ __align__(16) bf16 sWs1[320 * 64];        // 40 KB [K=320][N=64]
  __shared__ __align__(16) bf16 sWs2[64 * 256];        // 32 KB [K=64][N=256]
  __shared__ __align__(16) bf16 sRbf[32 * 320];        // 20 KB rbf tile
  __shared__ __align__(16) bf16 sH[32 * 64];           // 4 KB hidden tile
  __shared__ float sDist[256];
  __shared__ float sOut[256];

  const int tid  = threadIdx.x;
  const int lane = tid & 31;
  const int wave = tid >> 5;                           // 0..7
  const int b = blockIdx.x >> 8;
  const int i = blockIdx.x & 255;

  if (wave == 0) {                                     // TDM weight staging
    tdm_load_bf16(ldsAddrOf(sWs1), ws1b, 20480, 1, 20480);
    tdm_load_bf16(ldsAddrOf(sWs2), ws2b, 16384, 1, 16384);
    __builtin_amdgcn_s_wait_tensorcnt(0);
  }
  sDist[tid] = dist[(b * 256 + i) * 256 + tid];
  sOut[tid] = 0.f;
  __syncthreads();

  const int jt = wave >> 2;                            // 0..1 : j sub-tile
  const int w4 = wave & 3;                             // 0..3
  const int mHalf = (lane & 16) ? 8 : 0;
  const int lcol = lane & 15;

  const int rbfRow = tid & 31;
  const int rbfGrp = tid >> 5;                         // 8 groups x 40 k-values

  float acc[4] = {0.f, 0.f, 0.f, 0.f};

  for (int jc = 0; jc < 8; ++jc) {
    // ---- rbf tile: exp(-10*(d - 0.1k)^2), zero-padded k in [300,320)
    const float dj = sDist[jc * 32 + rbfRow];
#pragma unroll
    for (int c = 0; c < 40; ++c) {
      const int k = rbfGrp * 40 + c;
      const float x = dj - 0.1f * (float)k;
      const float e = (k < 300) ? __expf(-10.f * x * x) : 0.f;
      sRbf[rbfRow * 320 + k] = (bf16)e;
    }
    __syncthreads();

    // ---- GEMM1: h[16x16] tile per wave; K = 320 (10 k-tiles), all from LDS
    v8f hacc = {};
#pragma unroll
    for (int kt = 0; kt < 10; ++kt) {
      v16bf af = ldsFragA(sRbf + jt * 16 * 320 + kt * 32, 320, lane);
      v16bf bf_ = ldsFragB(sWs1 + kt * 32 * 64, 64, w4 * 16, lane);
      hacc = __builtin_amdgcn_wmma_f32_16x16x32_bf16(false, af, false, bf_,
                                                     (short)0, hacc, false, false);
    }
    {
      const float bn = bs1[w4 * 16 + lcol];
#pragma unroll
      for (int r = 0; r < 8; ++r) {
        const int m = r + mHalf;
        sH[(jt * 16 + m) * 64 + w4 * 16 + lcol] = (bf16)sspf(hacc[r] + bn);
      }
    }
    __syncthreads();

    // ---- GEMM2 + fold: each wave covers 4 column tiles (64 d-channels)
#pragma unroll
    for (int q = 0; q < 4; ++q) {
      const int nt = w4 * 4 + q;
      v8f dacc = {};
#pragma unroll
      for (int kt = 0; kt < 2; ++kt) {
        v16bf af = ldsFragA(sH + jt * 16 * 64 + kt * 32, 64, lane);
        v16bf bf_ = ldsFragB(sWs2 + kt * 32 * 256, 256, nt * 16, lane);
        dacc = __builtin_amdgcn_wmma_f32_16x16x32_bf16(false, af, false, bf_,
                                                       (short)0, dacc, false, false);
      }
      const float bn2 = bs2[nt * 16 + lcol];
      const int dcol = nt * 16 + lcol;
      const float* vrow = v1 + ((b * 256 + jc * 32 + jt * 16) * 256) + dcol;
#pragma unroll
      for (int r = 0; r < 8; ++r) {
        const int m = r + mHalf;                       // C/D layout: M = r + 8*(lane>=16)
        acc[q] += sspf(dacc[r] + bn2) * vrow[m * 256];
      }
    }
    __syncthreads();                                   // protect sRbf/sH
  }

  // ---- reduce: fold lane halves (column N lives in lanes N and N+16),
  //      then merge the two jt wave-groups via LDS float atomics.
#pragma unroll
  for (int q = 0; q < 4; ++q) {
    float vq = acc[q];
    vq += __shfl_xor(vq, 16, 32);
    if (lane < 16) atomicAdd(&sOut[(w4 * 4 + q) * 16 + lane], vq);
  }
  __syncthreads();
  u[(b * 256 + i) * 256 + tid] = sOut[tid];
}

// ---------------------------------------------------------------------------
extern "C" void kernel_launch(void* const* d_in, const int* in_sizes, int n_in,
                              void* d_out, int out_size, void* d_ws, size_t ws_size,
                              hipStream_t stream) {
  const float* v    = (const float*)d_in[0];   // [8,256,256]
  const float* dist = (const float*)d_in[1];   // [8,256,256]
  const float* mask = (const float*)d_in[2];   // [8,1,1,256] -> flat [2048]
  const float* Ws1  = (const float*)d_in[3];   // [300,64]
  const float* bs1  = (const float*)d_in[4];   // [64]
  const float* Ws2  = (const float*)d_in[5];   // [64,256]
  const float* bs2  = (const float*)d_in[6];   // [256]
  const float* Wa1  = (const float*)d_in[7];   // [256,256]
  const float* ba1  = (const float*)d_in[8];
  const float* Wa2  = (const float*)d_in[9];
  const float* ba2  = (const float*)d_in[10];
  const float* Wa3  = (const float*)d_in[11];
  const float* ba3  = (const float*)d_in[12];
  float* out = (float*)d_out;

  float* v1 = (float*)d_ws;                    // [2048,256] f32 (2 MB)
  float* uu = v1 + 2048 * 256;                 // [2048,256] f32 (2 MB)
  bf16* wa1b = (bf16*)(uu + 2048 * 256);       // bf16 weight images
  bf16* wa2b = wa1b + 65536;
  bf16* wa3b = wa2b + 65536;
  bf16* ws1b = wa3b + 65536;                   // [320*64] zero-padded
  bf16* ws2b = ws1b + 20480;                   // [64*256]

  k_prep<<<dim3(256), dim3(256), 0, stream>>>(Wa1, Wa2, Wa3, Ws1, Ws2,
                                              wa1b, wa2b, wa3b, ws1b, ws2b);
  // v1 = mask * ((mask*v) @ Wa1 + ba1)
  k_dense256<<<dim3(512), dim3(128), 0, stream>>>(v, wa1b, ba1, mask, v1, 1, 0, 1);
  // u[b,i,:] = sum_j ssp(ssp(rbf@Ws1+b)@Ws2+b) * v1[b,j,:]
  k_cfconv<<<dim3(2048), dim3(256), 0, stream>>>(v1, dist, ws1b, bs1, ws2b, bs2, uu);
  // v2 = mask * ssp((mask*u) @ Wa2 + ba2)   (reuse v1 buffer)
  k_dense256<<<dim3(512), dim3(128), 0, stream>>>(uu, wa2b, ba2, mask, v1, 1, 1, 1);
  // out = mask * (v2 @ Wa3 + ba3)
  k_dense256<<<dim3(512), dim3(128), 0, stream>>>(v1, wa3b, ba3, mask, out, 0, 0, 1);
}
